// Attention_54425825575314
// MI455X (gfx1250) — compile-verified, matrix operand order checked by reading, not run on hardware
//
#include <hip/hip_runtime.h>

typedef __attribute__((ext_vector_type(16))) _Float16 v16h;
typedef __attribute__((ext_vector_type(8)))  _Float16 v8h;
typedef __attribute__((ext_vector_type(8)))  float    v8f;

#define B_   32
#define T_   512
#define D_   512
#define L_   1024
#define M_   10
#define KTOT 1536   // 2D + D  (x_t | ctx | h)
#define NTOT 2048   // 4D gates
#define KT_  48     // KTOT / 32
#define NT_  128    // NTOT / 16
#define KSPLIT 4    // split-K factor for the gates GEMM (12 k-tiles each)
#define LSEG 8      // split-L factor for the attention GEMV (128 l each)

// ---------------------------------------------------------------------------
// Workspace layout (bytes):
//   Wp        fp16 packed weights [NT_][KT_][32][16]      0        6,291,456
//   bias      f32 [2048]                                  6,291,456    8,192
//   hS        f32 [32][512]                               6,299,648   65,536
//   cS        f32 [32][512]                               6,365,184   65,536
//   ctxPart   f32 [LSEG][32][512]                         6,430,720  524,288
//   Xh        fp16 [32][1536]                             6,955,008   98,304
//   gatesPart f32 [KSPLIT][32][2048]                      7,053,312 1,048,576
//   termrow   f32 [32][1024]                              8,101,888  131,072
// total ~7.85 MB
// ---------------------------------------------------------------------------

__device__ __forceinline__ float sigm_(float x)  { return 1.0f / (1.0f + __expf(-x)); }
__device__ __forceinline__ float tanh_s(float x) { return 1.0f - 2.0f / (__expf(2.0f * x) + 1.0f); }

// Pack W = [W_ih.T ; W_hh.T]  (K=1536, N=2048) into WMMA B-fragment layout.
// Fragment element i of lane: K = ktBase + (i<8 ? half*8 + i : 16 + half*8 + (i-8)),
// N = ntBase + (lane & 15), half = lane>>4.   (per CDNA5 ISA 16-bit 16x32 layout)
__global__ void pack_weights_k(const float* __restrict__ W_ih,
                               const float* __restrict__ W_hh,
                               _Float16* __restrict__ Wp) {
    int idx = blockIdx.x * blockDim.x + threadIdx.x;      // 0 .. 3,145,727
    if (idx >= NT_ * KT_ * 32 * 16) return;
    int i    = idx & 15;
    int lane = (idx >> 4) & 31;
    int tile = idx >> 9;
    int kt   = tile % KT_;
    int nt   = tile / KT_;
    int half = lane >> 4;
    int kin  = (i < 8) ? (half * 8 + i) : (16 + half * 8 + (i - 8));
    int k    = kt * 32 + kin;
    int n    = nt * 16 + (lane & 15);
    float v  = (k < 2 * D_) ? W_ih[(size_t)n * (2 * D_) + k]
                            : W_hh[(size_t)n * D_ + (k - 2 * D_)];
    Wp[idx] = (_Float16)v;
}

__global__ void setup_state_k(const float* __restrict__ b_ih, const float* __restrict__ b_hh,
                              float* __restrict__ bias, float* __restrict__ h,
                              float* __restrict__ c, float* __restrict__ ctxPart) {
    int idx = blockIdx.x * blockDim.x + threadIdx.x;
    if (idx < NTOT) bias[idx] = b_ih[idx] + b_hh[idx];
    if (idx < B_ * D_) { h[idx] = 0.f; c[idx] = 0.f; }
    if (idx < LSEG * B_ * D_) ctxPart[idx] = 0.f;
}

// Assemble X_t = [x_t | ctx | h] fp16 [32][1536]; ctx is summed from L-segments.
__global__ void pack_x_k(const float* __restrict__ xin, const float* __restrict__ ctxPart,
                         const float* __restrict__ h, _Float16* __restrict__ Xh, int t) {
    int idx = blockIdx.x * blockDim.x + threadIdx.x;
    if (idx >= B_ * KTOT) return;
    int b = idx / KTOT, k = idx % KTOT;
    float v;
    if (k < D_) {
        v = xin[((size_t)b * T_ + t) * D_ + k];
    } else if (k < 2 * D_) {
        int d = k - D_;
        v = 0.f;
#pragma unroll
        for (int s = 0; s < LSEG; ++s) v += ctxPart[(s * B_ + b) * D_ + d];
    } else {
        v = h[b * D_ + (k - 2 * D_)];
    }
    Xh[idx] = (_Float16)v;
}

// gatesPart[ks][32][2048] = X[:, ksK] @ W[ksK, :]  via v_wmma_f32_16x16x32_f16.
// 128 blocks (one N-tile each) x 4 waves (K segments): 512 waves across 128 WGPs.
__global__ void __launch_bounds__(128) gates_gemm_k(const _Float16* __restrict__ Xh,
                                                    const _Float16* __restrict__ Wp,
                                                    float* __restrict__ gatesPart) {
    int lane = threadIdx.x & 31;
    int ks   = threadIdx.x >> 5;          // 0..3 K-segment
    int nt   = blockIdx.x;                // 0..127 N-tile
    int half = lane >> 4;
    int m    = lane & 15;
    const int KSEG = KT_ / KSPLIT;        // 12 k-tiles

    v8f acc0 = {};  // rows 0-15
    v8f acc1 = {};  // rows 16-31
    const _Float16* wp = Wp + (size_t)nt * KT_ * 512 + (size_t)ks * KSEG * 512 + lane * 16;
    const _Float16* x0 = Xh + (size_t)m * KTOT + ks * KSEG * 32 + half * 8;
    const _Float16* x1 = Xh + (size_t)(16 + m) * KTOT + ks * KSEG * 32 + half * 8;

    for (int kt = 0; kt < KSEG; ++kt) {
        v16h bfrag = *(const v16h*)(wp);  wp += 512;
        v8h a0lo = *(const v8h*)(x0);
        v8h a0hi = *(const v8h*)(x0 + 16);
        v8h a1lo = *(const v8h*)(x1);
        v8h a1hi = *(const v8h*)(x1 + 16);
        x0 += 32; x1 += 32;
        v16h a0, a1;
#pragma unroll
        for (int i = 0; i < 8; ++i) {
            a0[i] = a0lo[i]; a0[8 + i] = a0hi[i];
            a1[i] = a1lo[i]; a1[8 + i] = a1hi[i];
        }
        acc0 = __builtin_amdgcn_wmma_f32_16x16x32_f16(false, a0, false, bfrag,
                                                      (short)0, acc0, false, false);
        acc1 = __builtin_amdgcn_wmma_f32_16x16x32_f16(false, a1, false, bfrag,
                                                      (short)0, acc1, false, false);
    }
    int n = nt * 16 + m;                   // C layout: col = lane&15
    float* gp = gatesPart + (size_t)ks * B_ * NTOT;
#pragma unroll
    for (int r = 0; r < 8; ++r) {          // C layout: lanes<16 -> M=r, lanes>=16 -> M=8+r
        gp[(size_t)(half * 8 + r) * NTOT + n]      = acc0[r];
        gp[(size_t)(16 + half * 8 + r) * NTOT + n] = acc1[r];
    }
}

__global__ void lstm_update_k(const float* __restrict__ gatesPart,
                              const float* __restrict__ bias,
                              float* __restrict__ h, float* __restrict__ c) {
    int idx = blockIdx.x * blockDim.x + threadIdx.x;
    if (idx >= B_ * D_) return;
    int b = idx / D_, d = idx % D_;
    float gi = bias[d], gf = bias[D_ + d], gg = bias[2 * D_ + d], go = bias[3 * D_ + d];
#pragma unroll
    for (int ks = 0; ks < KSPLIT; ++ks) {
        const float* g = gatesPart + ((size_t)ks * B_ + b) * NTOT;
        gi += g[d]; gf += g[D_ + d]; gg += g[2 * D_ + d]; go += g[3 * D_ + d];
    }
    float cn = sigm_(gf) * c[idx] + sigm_(gi) * tanh_s(gg);
    float hn = sigm_(go) * tanh_s(cn);
    c[idx] = cn;
    h[idx] = hn;
}

// 256 blocks: blockIdx = seg*32 + b.  Each block: phi (redundant, tiny) ->
// mixture params -> w for its 128-l segment (+ alignment/term) -> partial ctx GEMV.
__global__ void __launch_bounds__(256) attention_k(const float* __restrict__ h,
                                                   const float* __restrict__ W_g,
                                                   const float* __restrict__ b_g,
                                                   const float* __restrict__ memory,
                                                   float* __restrict__ ctxPart,
                                                   float* __restrict__ termrow,
                                                   float* __restrict__ align_out, int t) {
    __shared__ float sphi[32];
    __shared__ float sw[L_ / LSEG];
    __shared__ float salpha[M_], sksi[M_], sbinv[M_];
    int b     = blockIdx.x & (B_ - 1);
    int seg   = blockIdx.x >> 5;
    int lBase = seg * (L_ / LSEG);
    int tid   = threadIdx.x;
    int lane  = tid & 31;
    int wave  = tid >> 5;                       // 8 waves
    const float* hb = h + b * D_;

    // phi[j] = h . W_g[j] + b_g[j],  j = 0..29
    for (int j = wave; j < 3 * M_; j += 8) {
        const float* wg = W_g + (size_t)j * D_;
        float s = 0.f;
        for (int d2 = lane; d2 < D_; d2 += 32) s += hb[d2] * wg[d2];
        for (int off = 16; off; off >>= 1) s += __shfl_xor(s, off, 32);
        if (lane == 0) sphi[j] = s + b_g[j];
    }
    __syncthreads();

    if (tid == 0) {
        float mx = -1e30f;
        for (int m = 0; m < M_; ++m) mx = fmaxf(mx, sphi[2 * M_ + m]);
        float sum = 0.f;
        for (int m = 0; m < M_; ++m) { float e = __expf(sphi[2 * M_ + m] - mx); salpha[m] = e; sum += e; }
        float inv = 1.f / sum;
        for (int m = 0; m < M_; ++m) {
            salpha[m] *= inv;
            sksi[m]  = __expf(sphi[m]);          // ksi
            sbinv[m] = __expf(-sphi[M_ + m]);    // 1/beta
        }
    }
    __syncthreads();

    // w[l] = cdf(l+0.5) - cdf(l-0.5);  term = 1 - cdf(l+0.5)
    if (tid < L_ / LSEG) {
        int l = lBase + tid;
        float u = (float)l;
        float t1 = 0.f, t0 = 0.f;
#pragma unroll
        for (int m = 0; m < M_; ++m) {
            float a = salpha[m], ks = sksi[m], bi = sbinv[m];
            t1 += a * sigm_((u + 0.5f - ks) * bi);
            t0 += a * sigm_((u - 0.5f - ks) * bi);
        }
        float w = t1 - t0;
        sw[tid] = w;
        align_out[((size_t)b * T_ + t) * L_ + l] = w;
        termrow[b * L_ + l] = 1.0f - t1;
    }
    __syncthreads();

    // ctxPart[seg,b,d] = sum_{l in segment} w[l] * memory[b,l,d]
    const float* mb = memory + ((size_t)b * L_ + lBase) * D_;
    for (int d = tid; d < D_; d += 256) {
        const float* mp = mb + d;
        float acc = 0.f;
        for (int l0 = 0; l0 < L_ / LSEG; l0 += 8) {
            if (l0 + 8 < L_ / LSEG) __builtin_prefetch(mp + (size_t)(l0 + 8) * D_, 0, 1);
#pragma unroll
            for (int i = 0; i < 8; ++i) acc = fmaf(sw[l0 + i], mp[(size_t)(l0 + i) * D_], acc);
        }
        ctxPart[((size_t)seg * B_ + b) * D_ + d] = acc;
    }
}

__global__ void finalize_k(const float* __restrict__ ctxPart, const float* __restrict__ termrow,
                           const int* __restrict__ lengths, float* __restrict__ out) {
    int idx = blockIdx.x * blockDim.x + threadIdx.x;
    if (idx < B_ * D_) {
        float v = 0.f;
#pragma unroll
        for (int s = 0; s < LSEG; ++s) v += ctxPart[s * B_ * D_ + idx];
        out[idx] = v;
    }
    if (idx < B_) {
        int len = lengths[idx];
        if (len < 1) len = 1;
        if (len > L_) len = L_;
        out[(size_t)B_ * D_ + (size_t)B_ * T_ * L_ + idx] = termrow[idx * L_ + (len - 1)];
    }
}

extern "C" void kernel_launch(void* const* d_in, const int* in_sizes, int n_in,
                              void* d_out, int out_size, void* d_ws, size_t ws_size,
                              hipStream_t stream) {
    const float* input_h_c = (const float*)d_in[0];
    const float* memory    = (const float*)d_in[1];
    const int*   lengths   = (const int*)d_in[2];
    const float* W_ih      = (const float*)d_in[3];
    const float* W_hh      = (const float*)d_in[4];
    const float* b_ih      = (const float*)d_in[5];
    const float* b_hh      = (const float*)d_in[6];
    const float* W_g       = (const float*)d_in[7];
    const float* b_g       = (const float*)d_in[8];

    char* ws = (char*)d_ws;
    _Float16* Wp     = (_Float16*)(ws);
    float* bias      = (float*)(ws + 6291456);
    float* hS        = (float*)(ws + 6299648);
    float* cS        = (float*)(ws + 6365184);
    float* ctxPart   = (float*)(ws + 6430720);
    _Float16* Xh     = (_Float16*)(ws + 6955008);
    float* gatesPart = (float*)(ws + 7053312);
    float* termrow   = (float*)(ws + 8101888);

    float* out       = (float*)d_out;
    float* align_out = out + (size_t)B_ * D_;

    pack_weights_k<<<12288, 256, 0, stream>>>(W_ih, W_hh, Wp);
    setup_state_k<<<512, 256, 0, stream>>>(b_ih, b_hh, bias, hS, cS, ctxPart);

    for (int t = 0; t < T_; ++t) {
        pack_x_k<<<192, 256, 0, stream>>>(input_h_c, ctxPart, hS, Xh, t);
        gates_gemm_k<<<128, 128, 0, stream>>>(Xh, Wp, gatesPart);
        lstm_update_k<<<64, 256, 0, stream>>>(gatesPart, bias, hS, cS);
        attention_k<<<256, 256, 0, stream>>>(hS, W_g, b_g, memory, ctxPart, termrow,
                                             align_out, t);
    }
    finalize_k<<<64, 256, 0, stream>>>(ctxPart, termrow, lengths, out);
}
